// SelfAttention_10505490006224
// MI455X (gfx1250) — matrix operand order repaired and checked by reading, not compiled, not run on hardware
//
#include <hip/hip_runtime.h>
#include <hip/hip_bf16.h>
#include <math.h>

// ---------------------------------------------------------------------------
// Self-attention for MI455X (gfx1250): bf16 WMMA everywhere, wave32 layouts,
// 2-deep software pipelining so WMMAs overlap global loads (partial waits).
//   B=4, T=2048, C=1024, H=16, D=64  (hard-coded from the reference)
// ---------------------------------------------------------------------------

#define BATCH   4
#define SEQ     2048
#define DIM     1024
#define NHEAD   16
#define HDIM    64
#define MROWS   (BATCH * SEQ)          // 8192

typedef __bf16 bf16;
typedef __attribute__((ext_vector_type(16))) __bf16 v16bf;
typedef __attribute__((ext_vector_type(8)))  __bf16 v8bf;
typedef __attribute__((ext_vector_type(8)))  float  v8f;

union Frag { v16bf v; v8bf h[2]; };

__device__ __forceinline__ unsigned short bf16_bits(float f) {
    unsigned u = __float_as_uint(f);
    u += 0x7FFFu + ((u >> 16) & 1u);     // round-to-nearest-even
    return (unsigned short)(u >> 16);
}
__device__ __forceinline__ bf16 to_bf16(float f) {
    unsigned short s = bf16_bits(f);
    return __builtin_bit_cast(bf16, s);
}

__device__ __forceinline__ v8f wmma_bf16(const Frag& a, const Frag& b, v8f c) {
    // D = A(16x32 bf16) * B(32x16 bf16) + C(16x16 f32)
    return __builtin_amdgcn_wmma_f32_16x16x32_bf16(
        false, a.v, false, b.v, (short)0, c, false, false);
}

// --- A fragment: 16x32 bf16, row-major source, row stride ld (elements) ----
// lanes 0-15: row=lane,    K chunks [0..7]  and [16..23]
// lanes 16-31: row=lane-16, K chunks [8..15] and [24..31]
__device__ __forceinline__ void load_afrag(Frag& f, const bf16* base, int ld,
                                           int k0, int l16, int half) {
    const bf16* row = base + (size_t)l16 * ld + k0;
    f.h[0] = *(const v8bf*)(row + half * 8);
    f.h[1] = *(const v8bf*)(row + 16 + half * 8);
}

// --- B fragment: 32x16 bf16 from "transposed" storage srcT[n*ldn + k] ------
// lanes 0-15: col=lane,    K chunks [0..7] and [16..23] ; lanes 16-31: col=lane-16, K chunks [8..15] and [24..31]
__device__ __forceinline__ void load_bfrag(Frag& f, const bf16* srcT, int ldn,
                                           int k0, int l16, int half) {
    const bf16* row = srcT + (size_t)l16 * ldn + k0;
    f.h[0] = *(const v8bf*)(row + half * 8);
    f.h[1] = *(const v8bf*)(row + 16 + half * 8);
}

// ---------------------------------------------------------------------------
// Kernel 1: f32 -> bf16 cast (vectorized 4-wide)
// ---------------------------------------------------------------------------
__global__ void cast_f32_to_bf16(const float4* __restrict__ src,
                                 ushort4* __restrict__ dst, int n4) {
    int i = blockIdx.x * blockDim.x + threadIdx.x;
    int stride = gridDim.x * blockDim.x;
    for (; i < n4; i += stride) {
        float4 f = src[i];
        ushort4 o;
        o.x = bf16_bits(f.x); o.y = bf16_bits(f.y);
        o.z = bf16_bits(f.z); o.w = bf16_bits(f.w);
        dst[i] = o;
    }
}

// ---------------------------------------------------------------------------
// Kernel 2: fused QKV projection.  Each wave: one 16x16 tile of Q, K, V
// sharing the A fragment (3 WMMAs per 32-wide K step), 2-deep pipelined.
//   Q,K -> [B,H,T,D] bf16 ;  V -> [B,H,D,T] bf16 (transposed for PV B-frags)
// ---------------------------------------------------------------------------
struct QkvSet { Frag a, bq, bk, bv; };

__device__ __forceinline__ void qkv_load(QkvSet& f, const bf16* arow,
                                         const bf16* qrow, const bf16* krow,
                                         const bf16* vrow, int k,
                                         int l16, int half) {
    load_afrag(f.a,  arow, DIM, k, l16, half);
    load_bfrag(f.bq, qrow, DIM, k, l16, half);
    load_bfrag(f.bk, krow, DIM, k, l16, half);
    load_bfrag(f.bv, vrow, DIM, k, l16, half);
}

__global__ __launch_bounds__(128) void qkv_proj_kernel(
    const bf16* __restrict__ xb,
    const bf16* __restrict__ qw, const bf16* __restrict__ kw,
    const bf16* __restrict__ vw,
    const float* __restrict__ qb, const float* __restrict__ kb,
    const float* __restrict__ vb,
    bf16* __restrict__ Q, bf16* __restrict__ K, bf16* __restrict__ Vt)
{
    const int lane = threadIdx.x & 31;
    const int wave = threadIdx.x >> 5;
    const int l16  = lane & 15;
    const int half = lane >> 4;
    const int m0   = blockIdx.x * 16;
    const int n0   = (blockIdx.y * 4 + wave) * 16;

    v8f accq = {}, acck = {}, accv = {};
    const bf16* arow = xb + (size_t)m0 * DIM;
    const bf16* qrow = qw + (size_t)n0 * DIM;
    const bf16* krow = kw + (size_t)n0 * DIM;
    const bf16* vrow = vw + (size_t)n0 * DIM;

    QkvSet f0, f1;
    qkv_load(f0, arow, qrow, krow, vrow, 0, l16, half);

    #pragma unroll 1
    for (int k = 0; k < DIM - 64; k += 64) {
        qkv_load(f1, arow, qrow, krow, vrow, k + 32, l16, half);
        accq = wmma_bf16(f0.a, f0.bq, accq);
        acck = wmma_bf16(f0.a, f0.bk, acck);
        accv = wmma_bf16(f0.a, f0.bv, accv);
        qkv_load(f0, arow, qrow, krow, vrow, k + 64, l16, half);
        accq = wmma_bf16(f1.a, f1.bq, accq);
        acck = wmma_bf16(f1.a, f1.bk, acck);
        accv = wmma_bf16(f1.a, f1.bv, accv);
    }
    qkv_load(f1, arow, qrow, krow, vrow, DIM - 32, l16, half);
    accq = wmma_bf16(f0.a, f0.bq, accq);
    acck = wmma_bf16(f0.a, f0.bk, acck);
    accv = wmma_bf16(f0.a, f0.bv, accv);
    accq = wmma_bf16(f1.a, f1.bq, accq);
    acck = wmma_bf16(f1.a, f1.bk, acck);
    accv = wmma_bf16(f1.a, f1.bv, accv);

    const int n = n0 + l16;
    const int h = n >> 6, d = n & 63;
    const float bq_ = qb[n], bk_ = kb[n], bv_ = vb[n];
    #pragma unroll
    for (int i = 0; i < 8; ++i) {
        const int m = m0 + i + half * 8;
        const int b = m >> 11, t = m & (SEQ - 1);
        const size_t qk_off = (((size_t)(b * NHEAD + h)) * SEQ + t) * HDIM + d;
        Q[qk_off] = to_bf16(accq[i] + bq_);
        K[qk_off] = to_bf16(acck[i] + bk_);
        Vt[(((size_t)(b * NHEAD + h)) * HDIM + d) * SEQ + t] = to_bf16(accv[i] + bv_);
    }
}

// ---------------------------------------------------------------------------
// Kernel 3: causal flash attention.  grid(B*H, T/64) block(128).
// Each wave: 16 query rows.  Per 32-key step: 4 score WMMAs, then issue next
// K-frag + current V-frag loads, softmax VALU overlaps them, P relayout
// through per-wave LDS tile, 4 PV WMMAs.
// ---------------------------------------------------------------------------
__device__ __forceinline__ void load_kfrags(Frag kf[4], const bf16* Kh, int kbb,
                                            int l16, int half) {
    load_bfrag(kf[0], Kh + (size_t)kbb * HDIM,        HDIM, 0,  l16, half);
    load_bfrag(kf[1], Kh + (size_t)kbb * HDIM,        HDIM, 32, l16, half);
    load_bfrag(kf[2], Kh + (size_t)(kbb + 16) * HDIM, HDIM, 0,  l16, half);
    load_bfrag(kf[3], Kh + (size_t)(kbb + 16) * HDIM, HDIM, 32, l16, half);
}

__global__ __launch_bounds__(128) void flash_attn_kernel(
    const bf16* __restrict__ Q, const bf16* __restrict__ K,
    const bf16* __restrict__ Vt, bf16* __restrict__ Oa)
{
    __shared__ unsigned short s_p[4][16 * 32];   // per-wave P staging (1KB ea)

    const int lane = threadIdx.x & 31;
    const int wave = threadIdx.x >> 5;
    const int l16  = lane & 15;
    const int half = lane >> 4;
    const int bh    = blockIdx.x;                 // b*NHEAD + h
    const int qbase = blockIdx.y * 64 + wave * 16;
    const int qmax  = qbase + 15;

    const bf16* Qh = Q  + (size_t)bh * SEQ * HDIM;
    const bf16* Kh = K  + (size_t)bh * SEQ * HDIM;
    const bf16* Vh = Vt + (size_t)bh * HDIM * SEQ;

    // Q A-fragments for d=0..31 and d=32..63 (persist whole kernel)
    Frag qa0, qa1;
    load_afrag(qa0, Qh + (size_t)qbase * HDIM, HDIM, 0,  l16, half);
    load_afrag(qa1, Qh + (size_t)qbase * HDIM, HDIM, 32, l16, half);

    float mrow[8], lrow[8];
    #pragma unroll
    for (int i = 0; i < 8; ++i) { mrow[i] = -__builtin_inff(); lrow[i] = 0.0f; }
    v8f acc[4] = {{}, {}, {}, {}};

    const float sc = 0.125f * 1.4426950408889634f;   // 1/sqrt(D) * log2(e)

    Frag kcur[4];
    load_kfrags(kcur, Kh, 0, l16, half);

    int kbb = 0;
    while (true) {
        // ---- scores: S(16x32) = Q(16x64) . K^T, two 16-col WMMA tiles ----
        v8f s0 = {}, s1 = {};
        s0 = wmma_bf16(qa0, kcur[0], s0);
        s0 = wmma_bf16(qa1, kcur[1], s0);
        s1 = wmma_bf16(qa0, kcur[2], s1);
        s1 = wmma_bf16(qa1, kcur[3], s1);

        // ---- issue next K-block + current V-block loads; softmax overlaps --
        const int knext = kbb + 32;
        const bool more = (knext <= qmax);
        Frag knf[4];
        if (more) load_kfrags(knf, Kh, knext, l16, half);
        Frag vf[4];
        #pragma unroll
        for (int j = 0; j < 4; ++j)
            load_bfrag(vf[j], Vh + (size_t)(j * 16) * SEQ, SEQ, kbb, l16, half);

        // ---- scale (exp2 domain) + causal mask near diagonal ----
        if (kbb + 31 > qbase) {
            #pragma unroll
            for (int i = 0; i < 8; ++i) {
                const int qr = qbase + i + half * 8;
                const int j0 = kbb + l16, j1 = j0 + 16;
                s0[i] = (j0 <= qr) ? s0[i] * sc : -__builtin_inff();
                s1[i] = (j1 <= qr) ? s1[i] * sc : -__builtin_inff();
            }
        } else {
            #pragma unroll
            for (int i = 0; i < 8; ++i) { s0[i] *= sc; s1[i] *= sc; }
        }

        // ---- online softmax (row = 16-lane half of the wave) ----
        #pragma unroll
        for (int i = 0; i < 8; ++i) {
            float mx = fmaxf(s0[i], s1[i]);
            #pragma unroll
            for (int off = 8; off; off >>= 1)
                mx = fmaxf(mx, __shfl_xor(mx, off, 32));
            const float mnew = fmaxf(mrow[i], mx);
            const float fac  = __builtin_exp2f(mrow[i] - mnew);
            mrow[i] = mnew;
            const float p0 = __builtin_exp2f(s0[i] - mnew);
            const float p1 = __builtin_exp2f(s1[i] - mnew);
            s0[i] = p0; s1[i] = p1;
            float sum = p0 + p1;
            #pragma unroll
            for (int off = 8; off; off >>= 1)
                sum += __shfl_xor(sum, off, 32);
            lrow[i] = lrow[i] * fac + sum;
            acc[0][i] *= fac; acc[1][i] *= fac;
            acc[2][i] *= fac; acc[3][i] *= fac;
        }

        // ---- P: C-layout -> LDS -> A-layout (per-wave, DS is in-order) ----
        unsigned short* sp = s_p[wave];
        #pragma unroll
        for (int i = 0; i < 8; ++i) {
            const int r = i + half * 8;
            sp[r * 32 + l16]      = bf16_bits(s0[i]);
            sp[r * 32 + 16 + l16] = bf16_bits(s1[i]);
        }
        Frag pa;
        const unsigned short* prow = sp + l16 * 32;
        pa.h[0] = *(const v8bf*)(prow + half * 8);
        pa.h[1] = *(const v8bf*)(prow + 16 + half * 8);

        // ---- O += P(16x32) . V(32x64) ----
        #pragma unroll
        for (int j = 0; j < 4; ++j)
            acc[j] = wmma_bf16(pa, vf[j], acc[j]);

        if (!more) break;
        #pragma unroll
        for (int j = 0; j < 4; ++j) kcur[j] = knf[j];
        kbb = knext;
    }

    // ---- epilogue: normalize, write [B, T, H*D] bf16 for the O-proj ----
    const int b = bh >> 4, h = bh & 15;
    #pragma unroll
    for (int i = 0; i < 8; ++i) {
        const float inv = 1.0f / lrow[i];
        const int q = qbase + i + half * 8;
        const size_t orow = ((size_t)(b * SEQ + q)) * DIM + h * HDIM;
        #pragma unroll
        for (int j = 0; j < 4; ++j)
            Oa[orow + j * 16 + l16] = to_bf16(acc[j][i] * inv);
    }
}

// ---------------------------------------------------------------------------
// Kernel 4: output projection.  Each wave: 16x64 strip (A-frag reused 4x),
// 2-deep pipelined.  f32 output + bias -> d_out.
// ---------------------------------------------------------------------------
struct OprojSet { Frag a, b[4]; };

__device__ __forceinline__ void oproj_load(OprojSet& f, const bf16* arow,
                                           const bf16* wbase, int k,
                                           int l16, int half) {
    load_afrag(f.a, arow, DIM, k, l16, half);
    #pragma unroll
    for (int j = 0; j < 4; ++j)
        load_bfrag(f.b[j], wbase + (size_t)(j * 16) * DIM, DIM, k, l16, half);
}

__global__ __launch_bounds__(128) void oproj_kernel(
    const bf16* __restrict__ A, const bf16* __restrict__ Wt,
    const float* __restrict__ bias, float* __restrict__ out)
{
    const int lane = threadIdx.x & 31;
    const int wave = threadIdx.x >> 5;
    const int l16  = lane & 15;
    const int half = lane >> 4;
    const int m0 = (blockIdx.x * 4 + wave) * 16;
    const int n0 = blockIdx.y * 64;

    v8f acc[4] = {{}, {}, {}, {}};
    const bf16* arow  = A + (size_t)m0 * DIM;
    const bf16* wbase = Wt + (size_t)n0 * DIM;

    OprojSet f0, f1;
    oproj_load(f0, arow, wbase, 0, l16, half);

    #pragma unroll 1
    for (int k = 0; k < DIM - 64; k += 64) {
        oproj_load(f1, arow, wbase, k + 32, l16, half);
        #pragma unroll
        for (int j = 0; j < 4; ++j) acc[j] = wmma_bf16(f0.a, f0.b[j], acc[j]);
        oproj_load(f0, arow, wbase, k + 64, l16, half);
        #pragma unroll
        for (int j = 0; j < 4; ++j) acc[j] = wmma_bf16(f1.a, f1.b[j], acc[j]);
    }
    oproj_load(f1, arow, wbase, DIM - 32, l16, half);
    #pragma unroll
    for (int j = 0; j < 4; ++j) acc[j] = wmma_bf16(f0.a, f0.b[j], acc[j]);
    #pragma unroll
    for (int j = 0; j < 4; ++j) acc[j] = wmma_bf16(f1.a, f1.b[j], acc[j]);

    #pragma unroll
    for (int i = 0; i < 8; ++i) {
        const int m = m0 + i + half * 8;
        #pragma unroll
        for (int j = 0; j < 4; ++j) {
            const int n = n0 + j * 16 + l16;
            out[(size_t)m * DIM + n] = acc[j][i] + bias[n];
        }
    }
}

// ---------------------------------------------------------------------------
// Host launcher
// ---------------------------------------------------------------------------
extern "C" void kernel_launch(void* const* d_in, const int* in_sizes, int n_in,
                              void* d_out, int out_size, void* d_ws, size_t ws_size,
                              hipStream_t stream) {
    const float* x   = (const float*)d_in[0];
    const float* q_w = (const float*)d_in[1];
    const float* q_b = (const float*)d_in[2];
    const float* k_w = (const float*)d_in[3];
    const float* k_b = (const float*)d_in[4];
    const float* v_w = (const float*)d_in[5];
    const float* v_b = (const float*)d_in[6];
    const float* o_w = (const float*)d_in[7];
    const float* o_b = (const float*)d_in[8];

    char* ws = (char*)d_ws;
    const size_t MB = 1u << 20;
    bf16* xb  = (bf16*)(ws + 0 * MB);     // 16 MB  x in bf16
    bf16* qwb = (bf16*)(ws + 16 * MB);    //  2 MB
    bf16* kwb = (bf16*)(ws + 18 * MB);    //  2 MB
    bf16* vwb = (bf16*)(ws + 20 * MB);    //  2 MB
    bf16* owb = (bf16*)(ws + 22 * MB);    //  2 MB
    bf16* Qb  = (bf16*)(ws + 24 * MB);    // 16 MB  [B,H,T,D]
    bf16* Kb  = (bf16*)(ws + 40 * MB);    // 16 MB  [B,H,T,D]
    bf16* Vtb = (bf16*)(ws + 56 * MB);    // 16 MB  [B,H,D,T]
    bf16* Aob = (bf16*)(ws + 72 * MB);    // 16 MB  attention out [B*T, H*D]

    // 1) casts to bf16
    {
        const int nx4 = (MROWS * DIM) / 4;      // 2,097,152
        const int nw4 = (DIM * DIM) / 4;        //   262,144
        cast_f32_to_bf16<<<(nx4 + 255) / 256, 256, 0, stream>>>(
            (const float4*)x, (ushort4*)xb, nx4);
        cast_f32_to_bf16<<<(nw4 + 255) / 256, 256, 0, stream>>>(
            (const float4*)q_w, (ushort4*)qwb, nw4);
        cast_f32_to_bf16<<<(nw4 + 255) / 256, 256, 0, stream>>>(
            (const float4*)k_w, (ushort4*)kwb, nw4);
        cast_f32_to_bf16<<<(nw4 + 255) / 256, 256, 0, stream>>>(
            (const float4*)v_w, (ushort4*)vwb, nw4);
        cast_f32_to_bf16<<<(nw4 + 255) / 256, 256, 0, stream>>>(
            (const float4*)o_w, (ushort4*)owb, nw4);
    }

    // 2) fused QKV projection
    qkv_proj_kernel<<<dim3(MROWS / 16, DIM / 64), 128, 0, stream>>>(
        xb, qwb, kwb, vwb, q_b, k_b, v_b, Qb, Kb, Vtb);

    // 3) causal flash attention
    flash_attn_kernel<<<dim3(BATCH * NHEAD, SEQ / 64), 128, 0, stream>>>(
        Qb, Kb, Vtb, Aob);

    // 4) output projection -> f32 d_out
    oproj_kernel<<<dim3(MROWS / 64, DIM / 64), 128, 0, stream>>>(
        Aob, owb, o_b, (float*)d_out);
}
